// UnionAttention_6691559047252
// MI455X (gfx1250) — matrix operand order repaired and checked by reading, compiled, not run on hardware
//
#include <hip/hip_runtime.h>
#include <hip/hip_bf16.h>

typedef __attribute__((ext_vector_type(16))) _Float16 v16h;
typedef __attribute__((ext_vector_type(8)))  float    v8f;

#define EPSBN 1e-3f
#define NPIX  6400          // H*W
#define CHUNK 640           // 6400/640 = 10 chunks; 640/64 = 10 double wmma-steps/chunk

// ---- workspace layout (float offsets) ----
#define PARTOFF 0                   // 64 blocks * 128 partial channel sums
#define WPROJ   8192                // 384: folded proj weights [(d*2+b)*64 + c]
#define CQKV    8576                // 6:   constant terms      [d*2+b]
#define KMM     8582                // 4 uints: [0..1]=kmax_enc[b], [2..3]=kmin_enc[b]
#define QFO     8592                // 12800
#define KFO     (QFO + 12800)       // 12800
#define VFO     (KFO + 12800)       // 12800  (end = 46992 floats ~ 188 KB)

// monotonic float<->uint encoding so integer atomicMin/Max order like floats
__device__ __forceinline__ unsigned fenc(float f) {
  unsigned u = __float_as_uint(f);
  return (u >> 31) ? ~u : (u | 0x80000000u);
}
__device__ __forceinline__ float fdec(unsigned e) {
  unsigned u = (e >> 31) ? (e & 0x7fffffffu) : ~e;
  return __uint_as_float(u);
}

// ---------------- K1: per-channel partial sums of x (for GAP) ----------------
__global__ __launch_bounds__(256) void k_gap_partial(const float* __restrict__ x,
                                                     float* __restrict__ ws) {
  __shared__ float t0[256], t1[256];
  const int tid = threadIdx.x;
  float a0 = 0.f, a1 = 0.f;
  // stride (64*256) is a multiple of 64 -> each thread touches a fixed channel c = tid&63
  for (int i = blockIdx.x * 256 + tid; i < 2 * NPIX * 64; i += 64 * 256) {
    float v = x[i];
    if (i < NPIX * 64) a0 += v; else a1 += v;
  }
  t0[tid] = a0; t1[tid] = a1;
  __syncthreads();
  if (tid < 64) {  // deterministic 4-way fold: threads tid, tid+64, ... share channel c=tid
    float s0 = t0[tid] + t0[tid + 64] + t0[tid + 128] + t0[tid + 192];
    float s1 = t1[tid] + t1[tid + 64] + t1[tid + 128] + t1[tid + 192];
    ws[PARTOFF + blockIdx.x * 128 + tid]      = s0;
    ws[PARTOFF + blockIdx.x * 128 + 64 + tid] = s1;
  }
}

// --------- K2: channel attention + percentile mask + weight folding ---------
__global__ __launch_bounds__(256) void k_channel(
    const float* __restrict__ g1, const float* __restrict__ b1,
    const float* __restrict__ m1, const float* __restrict__ v1,
    const float* __restrict__ wqkvch, const float* __restrict__ wfc,
    const float* __restrict__ bfc,
    const float* __restrict__ g2, const float* __restrict__ b2,
    const float* __restrict__ m2, const float* __restrict__ v2,
    const float* __restrict__ wsp, float* ws) {
  __shared__ float sGN[128], sQKV[384], sATT[128], sA2[128], sTHR[2], sBW[384];
  const int tid = threadIdx.x;

  if (tid < 128) {                          // GAP + BN1
    const int c = tid & 63;
    float s = 0.f;
    for (int blk = 0; blk < 64; ++blk) s += ws[PARTOFF + blk * 128 + tid];
    float gap = s * (1.0f / 6400.0f);
    sGN[tid] = (gap - m1[c]) * rsqrtf(v1[c] + EPSBN) * g1[c] + b1[c];
  }
  __syncthreads();
  for (int idx = tid; idx < 384; idx += 256) {   // qkv = gn @ w_qkv_ch
    const int b = idx / 192, d = idx % 192;
    float s = 0.f;
    for (int c = 0; c < 64; ++c) s += sGN[b * 64 + c] * wqkvch[c * 192 + d];
    sQKV[b * 192 + d] = s;
  }
  __syncthreads();
  if (tid < 128) {                          // softmax(q_i * k_j) @ v   (64x64)
    const int b = tid >> 6, i = tid & 63;
    const float qi = sQKV[b * 192 + i];
    float mx = -3.4e38f;
    for (int j = 0; j < 64; ++j) mx = fmaxf(mx, qi * sQKV[b * 192 + 64 + j]);
    float S = 0.f, T = 0.f;
    for (int j = 0; j < 64; ++j) {
      float e = __expf(qi * sQKV[b * 192 + 64 + j] - mx);
      S += e; T += e * sQKV[b * 192 + 128 + j];
    }
    sATT[tid] = T / S;
  }
  __syncthreads();
  if (tid < 128) {                          // sigmoid(att @ w_fc + b_fc)
    const int b = tid >> 6, c = tid & 63;
    float s = bfc[c];
    for (int i = 0; i < 64; ++i) s += sATT[b * 64 + i] * wfc[i * 64 + c];
    sA2[tid] = 1.f / (1.f + __expf(-s));
  }
  __syncthreads();
  if (tid < 2) {                            // 10th percentile, linear interp (n=64)
    float arr[64];
    for (int i = 0; i < 64; ++i) arr[i] = sA2[tid * 64 + i];
    for (int i = 1; i < 64; ++i) {          // insertion sort, tiny
      float key = arr[i]; int j = i - 1;
      while (j >= 0 && arr[j] > key) { arr[j + 1] = arr[j]; --j; }
      arr[j + 1] = key;
    }
    sTHR[tid] = arr[6] + 0.3f * (arr[7] - arr[6]);   // pos = 0.1*63 = 6.3
  }
  __syncthreads();
  if (tid < 128) {                          // fold BN2 + mask + (1+att) into affine
    const int b = tid >> 6, c = tid & 63;
    const float a2 = sA2[tid];
    const float mk = (a2 > sTHR[b]) ? a2 : 0.f;
    const float inv = rsqrtf(v2[c] + EPSBN);
    const float gs = inv * g2[c];
    const float A  = (1.f + a2) * gs * mk;
    const float Bc = (b2[c] - m2[c] * gs) * mk;
    for (int d = 0; d < 3; ++d) {
      const float w = wsp[c * 3 + d];
      ws[WPROJ + (d * 2 + b) * 64 + c] = A * w;
      sBW[(d * 2 + b) * 64 + c] = Bc * w;
    }
  }
  __syncthreads();
  if (tid < 6) {                            // constant terms
    float s = 0.f;
    for (int c = 0; c < 64; ++c) s += sBW[tid * 64 + c];
    ws[CQKV + tid] = s;
  }
  if (tid == 6) {                           // init min/max cells (sortable encoding)
    unsigned* kmm = (unsigned*)(ws + KMM);
    kmm[0] = 0u; kmm[1] = 0u; kmm[2] = 0xFFFFFFFFu; kmm[3] = 0xFFFFFFFFu;
  }
}

// ------------- K3: fused spatial qkv projection + global k min/max -----------
__global__ __launch_bounds__(256) void k_qkv_sp(const float* __restrict__ x, float* ws) {
  __shared__ float sW[192];
  __shared__ float rmn[256], rmx[256];
  const int tid = threadIdx.x;
  const int b = blockIdx.x / 25;                     // 25 blocks per batch
  const int p = blockIdx.x * 256 + tid;              // global pixel index
  if (tid < 192) {
    const int d = tid / 64, c = tid & 63;
    sW[tid] = ws[WPROJ + (d * 2 + b) * 64 + c];
  }
  __syncthreads();
  float aq = ws[CQKV + 0 + b], ak = ws[CQKV + 2 + b], av = ws[CQKV + 4 + b];
  const float4* xr = (const float4*)(x + (size_t)p * 64);
#pragma unroll
  for (int i = 0; i < 16; ++i) {
    const float4 xv = xr[i];
    aq += xv.x * sW[4*i]      + xv.y * sW[4*i+1]      + xv.z * sW[4*i+2]      + xv.w * sW[4*i+3];
    ak += xv.x * sW[64+4*i]   + xv.y * sW[64+4*i+1]   + xv.z * sW[64+4*i+2]   + xv.w * sW[64+4*i+3];
    av += xv.x * sW[128+4*i]  + xv.y * sW[128+4*i+1]  + xv.z * sW[128+4*i+2]  + xv.w * sW[128+4*i+3];
  }
  ws[QFO + p] = aq; ws[KFO + p] = ak; ws[VFO + p] = av;
  rmn[tid] = ak; rmx[tid] = ak;
  __syncthreads();
  for (int s = 128; s > 0; s >>= 1) {
    if (tid < s) {
      rmn[tid] = fminf(rmn[tid], rmn[tid + s]);
      rmx[tid] = fmaxf(rmx[tid], rmx[tid + s]);
    }
    __syncthreads();
  }
  if (tid == 0) {
    unsigned* kmm = (unsigned*)(ws + KMM);
    atomicMax(&kmm[b],     fenc(rmx[0]));
    atomicMin(&kmm[2 + b], fenc(rmn[0]));
  }
}

// ---- K4: rank-1 softmax attention; exp tiles reduced against [v|1] by WMMA ----
// Two independent WMMA accumulator chains (even/odd 32-j slices) so the matrix
// pipe overlaps the TRANS/VALU exp+pack stream of the other chain.
__global__ __launch_bounds__(256) void k_spatial_attn(const float* __restrict__ ws,
                                                      float* __restrict__ out) {
  __shared__ float sk[CHUNK], sv[CHUNK];
  const int tid = threadIdx.x;
  const int b = blockIdx.x / 50;                     // 50 blocks per batch
  const int tileRow = (blockIdx.x % 50) * 128;       // 8 waves * 16 rows
  const int wave = tid >> 5, lane = tid & 31;
  const int rowBase = tileRow + wave * 16;

  const unsigned* kmm = (const unsigned*)(ws + KMM);
  const float kmax = fdec(kmm[b]), kmin = fdec(kmm[2 + b]);
  const float q = ws[QFO + b * NPIX + rowBase + (lane & 15)];   // A-row for this lane
  const float m = (q >= 0.f) ? q * kmax : q * kmin;             // exact row max of q*k

  // exp(x) = exp2(x*log2e): fold log2e into the per-row constants
  const float LOG2E = 1.4426950408889634f;
  const float qs = q * LOG2E;
  const float ms = m * LOG2E;

  v8f acc0 = {0.f, 0.f, 0.f, 0.f, 0.f, 0.f, 0.f, 0.f};
  v8f acc1 = {0.f, 0.f, 0.f, 0.f, 0.f, 0.f, 0.f, 0.f};
  // 16-bit A 16x32 layout: lanes 0-15 row=lane, K={0..7,16..23}; lanes 16-31 row=lane-16, K={8..15,24..31}
  const int off0 = (lane < 16) ? 0 : 8;
  const int off1 = (lane < 16) ? 16 : 24;
  const int rowoff = (lane < 16) ? 0 : 8;            // B rows {0..7,16..23} vs {8..15,24..31}
  // branchless B-operand column select: col0 = v (mV=1), col1 = ones (mOne=1), else 0
  const float mV   = ((lane & 15) == 0) ? 1.f : 0.f;
  const float mOne = ((lane & 15) == 1) ? 1.f : 0.f;

  for (int c0 = 0; c0 < NPIX; c0 += CHUNK) {
    __syncthreads();
    for (int i = tid; i < CHUNK; i += 256) {
      sk[i] = ws[KFO + b * NPIX + c0 + i];
      sv[i] = ws[VFO + b * NPIX + c0 + i];
    }
    __syncthreads();
    if (c0 + CHUNK < NPIX && tid * 4 < CHUNK) {      // hint next chunk into cache
      __builtin_prefetch(&ws[KFO + b * NPIX + c0 + CHUNK + tid * 4], 0, 1);
      __builtin_prefetch(&ws[VFO + b * NPIX + c0 + CHUNK + tid * 4], 0, 1);
    }
    for (int jj = 0; jj < CHUNK; jj += 64) {
#pragma unroll
      for (int half = 0; half < 2; ++half) {
        const int j0 = jj + half * 32;
        v16h a;
#pragma unroll
        for (int t = 0; t < 8; ++t) {
          float e = __builtin_amdgcn_exp2f(__builtin_fmaf(qs, sk[j0 + off0 + t], -ms));
          a[t] = (_Float16)e;
        }
#pragma unroll
        for (int t = 0; t < 8; ++t) {
          float e = __builtin_amdgcn_exp2f(__builtin_fmaf(qs, sk[j0 + off1 + t], -ms));
          a[8 + t] = (_Float16)e;
        }
        v16h bm;                                     // branchless: all lanes read LDS
#pragma unroll
        for (int i = 0; i < 8; ++i) {
          float lo = __builtin_fmaf(sv[j0 + rowoff + ((2 * i) & 7) + 16 * (i >> 2)],     mV, mOne);
          float hi = __builtin_fmaf(sv[j0 + rowoff + ((2 * i) & 7) + 16 * (i >> 2) + 1], mV, mOne);
          bm[2 * i]     = (_Float16)lo;
          bm[2 * i + 1] = (_Float16)hi;
        }
        // D[M,0] += sum_j e * v ; D[M,1] += sum_j e
        if (half == 0)
          acc0 = __builtin_amdgcn_wmma_f32_16x16x32_f16(
              false, a, false, bm, (short)0, acc0, false, false);
        else
          acc1 = __builtin_amdgcn_wmma_f32_16x16x32_f16(
              false, a, false, bm, (short)0, acc1, false, false);
      }
    }
  }
  v8f acc = acc0 + acc1;

  // C/D layout: VGPR r: lanes 0-15 -> (M=r, N=lane); lanes 16-31 -> (M=8+r, N=lane-16)
  float outv = 0.f;
#pragma unroll
  for (int r = 0; r < 8; ++r) {
    float Tlo = __shfl(acc[r], 0, 32);
    float Slo = __shfl(acc[r], 1, 32);
    float Thi = __shfl(acc[r], 16, 32);
    float Shi = __shfl(acc[r], 17, 32);
    if (lane == r)     outv = Tlo / Slo;   // row rowBase + r
    if (lane == 8 + r) outv = Thi / Shi;   // row rowBase + 8 + r
  }
  if (lane < 16) {
    const int row = rowBase + lane;
    out[b * NPIX + row] = ws[VFO + b * NPIX + row] + outv;   // v2s + attn
  }
}

extern "C" void kernel_launch(void* const* d_in, const int* in_sizes, int n_in,
                              void* d_out, int out_size, void* d_ws, size_t ws_size,
                              hipStream_t stream) {
  (void)in_sizes; (void)n_in; (void)out_size; (void)ws_size;
  const float* x      = (const float*)d_in[0];
  const float* wqkvch = (const float*)d_in[1];
  const float* g1     = (const float*)d_in[2];
  const float* b1     = (const float*)d_in[3];
  const float* m1     = (const float*)d_in[4];
  const float* v1     = (const float*)d_in[5];
  const float* wfc    = (const float*)d_in[6];
  const float* bfc    = (const float*)d_in[7];
  const float* g2     = (const float*)d_in[8];
  const float* b2     = (const float*)d_in[9];
  const float* m2     = (const float*)d_in[10];
  const float* v2     = (const float*)d_in[11];
  const float* wsp    = (const float*)d_in[12];
  float* ws  = (float*)d_ws;
  float* out = (float*)d_out;

  k_gap_partial<<<64, 256, 0, stream>>>(x, ws);
  k_channel<<<1, 256, 0, stream>>>(g1, b1, m1, v1, wqkvch, wfc, bfc,
                                   g2, b2, m2, v2, wsp, ws);
  k_qkv_sp<<<50, 256, 0, stream>>>(x, ws);
  k_spatial_attn<<<100, 256, 0, stream>>>(ws, out);
}
